// E28ConvElmanCell_68556267978940
// MI455X (gfx1250) — matrix operand order repaired and checked by reading, not compiled, not run on hardware
//
#include <hip/hip_runtime.h>
#include <hip/hip_bf16.h>
#include <math.h>

#define Bn 8
#define Tn 2048
#define Dn 1024
#define Kc 4

typedef __attribute__((ext_vector_type(16))) __bf16 v16bf;
typedef __attribute__((ext_vector_type(8)))  __bf16 v8bf;
typedef __attribute__((ext_vector_type(4)))  __bf16 v4bf;
typedef __attribute__((ext_vector_type(8)))  float  v8f;
typedef __attribute__((ext_vector_type(4)))  float  v4f;

typedef __attribute__((ext_vector_type(4))) unsigned tdm_u32x4;
typedef __attribute__((ext_vector_type(8))) int      tdm_i32x8;
typedef __attribute__((ext_vector_type(4))) int      tdm_i32x4;

union ABFrag { v16bf v; v8bf h[2]; };

// A-matrix fragment (16x32 bf16): lane holds row (lane&15);
// elements 0..15 = K[hi*16 .. +15] (contiguous, matches B fragments).
__device__ __forceinline__ v16bf load_a_frag(const __bf16* rowk0, int hi) {
    ABFrag u;
    u.h[0] = *(const v8bf*)(rowk0 + hi * 16);
    u.h[1] = *(const v8bf*)(rowk0 + hi * 16 + 8);
    return u.v;
}

__device__ __forceinline__ void grid_sync(unsigned* cnt, unsigned* gen, unsigned nblocks) {
    __threadfence();
    __syncthreads();
    if (threadIdx.x == 0) {
        unsigned g = __hip_atomic_load(gen, __ATOMIC_RELAXED, __HIP_MEMORY_SCOPE_AGENT);
        unsigned arrived = __hip_atomic_fetch_add(cnt, 1u, __ATOMIC_ACQ_REL, __HIP_MEMORY_SCOPE_AGENT);
        if (arrived == nblocks - 1u) {
            __hip_atomic_store(cnt, 0u, __ATOMIC_RELAXED, __HIP_MEMORY_SCOPE_AGENT);
            __hip_atomic_store(gen, g + 1u, __ATOMIC_RELEASE, __HIP_MEMORY_SCOPE_AGENT);
        } else {
            while (__hip_atomic_load(gen, __ATOMIC_ACQUIRE, __HIP_MEMORY_SCOPE_AGENT) == g) {
                __builtin_amdgcn_s_sleep(2);
            }
        }
    }
    __syncthreads();
}

// ---- TDM: 1-D tile load, `nelem` bf16 elements from contiguous global -> LDS offset ----
#if __has_builtin(__builtin_amdgcn_tensor_load_to_lds)
#define HAVE_TDM 1
__device__ __forceinline__ void tdm_load_1d(const __bf16* gsrc, unsigned lds_off, unsigned nelem) {
    unsigned long long ga = (unsigned long long)(uintptr_t)gsrc;
    tdm_u32x4 g0;
    g0[0] = 1u;                                               // count=1, user mode
    g0[1] = lds_off;                                          // lds_addr (bytes)
    g0[2] = (unsigned)(ga & 0xFFFFFFFFu);                     // global_addr[31:0]
    g0[3] = (unsigned)((ga >> 32) & 0x1FFFFFFu) | (2u << 30); // global_addr[56:32], type=2
    tdm_i32x8 g1;
    g1[0] = 0x10000;                          // workgroup_mask=0, data_size=1 (2 bytes)
    g1[1] = (int)((nelem & 0xFFFFu) << 16);   // tensor_dim0[15:0] in bits[63:48]
    g1[2] = (int)(((nelem >> 16) & 0xFFFFu) | (1u << 16)); // tensor_dim0[31:16], tensor_dim1=1
    g1[3] = (int)((nelem & 0xFFFFu) << 16);   // tile_dim0 (16-bit) in bits[127:112]
    g1[4] = 1;                                // tile_dim1=1, tile_dim2=0
    g1[5] = (int)nelem;                       // tensor_dim0_stride[31:0]
    g1[6] = 0;
    g1[7] = 0;
    tdm_i32x4 gz = {0, 0, 0, 0};
#if __clang_major__ >= 23
    tdm_i32x8 gz8 = {0, 0, 0, 0, 0, 0, 0, 0};
    __builtin_amdgcn_tensor_load_to_lds(g0, g1, gz, gz, gz8, 0);
#else
    __builtin_amdgcn_tensor_load_to_lds(g0, g1, gz, gz, 0);
#endif
}
#endif

// ---- K0: convert weights to bf16, init h double-buffer (16 rows, 8..15 zero), reset sync ----
__global__ void k_setup(const float* __restrict__ Wx, const float* __restrict__ Wh,
                        const float* __restrict__ h_init,
                        __bf16* __restrict__ Wxbf, __bf16* __restrict__ Whbf,
                        __bf16* __restrict__ hbuf, unsigned* __restrict__ sync) {
    int idx = blockIdx.x * blockDim.x + threadIdx.x;
    if (idx < Dn * Dn) {
        Wxbf[idx] = (__bf16)Wx[idx];
        Whbf[idx] = (__bf16)Wh[idx];
    }
    if (idx < 16 * Dn) {
        int r = idx / Dn;
        __bf16 v = (__bf16)((r < Bn) ? h_init[idx] : 0.0f);
        hbuf[idx] = v;
        hbuf[16 * Dn + idx] = v;
    }
    if (idx < 2) sync[idx] = 0u;
}

// ---- K1: causal depthwise conv (K=4) + SiLU, bf16 A-matrix [B*T, D]. 4 channels/thread. ----
__global__ void k_conv(const float* __restrict__ x, const float* __restrict__ cw,
                       const float* __restrict__ cb, __bf16* __restrict__ Abf) {
    long e0 = ((long)blockIdx.x * blockDim.x + threadIdx.x) * 4;  // covers B*T*D
    int d0 = (int)(e0 % Dn);
    int t  = (int)((e0 / Dn) % Tn);
    v4f w[4];                                   // w[i] = 4 taps of channel d0+i
#pragma unroll
    for (int i = 0; i < 4; ++i) w[i] = *(const v4f*)(cw + (d0 + i) * Kc);
    v4f xv[Kc];
#pragma unroll
    for (int j = 0; j < Kc; ++j) {
        int tt = t - (Kc - 1) + j;
        v4f zero = {0.0f, 0.0f, 0.0f, 0.0f};
        xv[j] = (tt >= 0) ? *(const v4f*)(x + e0 + (long)(j - (Kc - 1)) * Dn) : zero;
    }
    v4bf o;
#pragma unroll
    for (int i = 0; i < 4; ++i) {
        float acc = cb[d0 + i];
#pragma unroll
        for (int j = 0; j < Kc; ++j) acc = fmaf(w[i][j], xv[j][i], acc);
        float s = acc / (1.0f + __expf(-acc));  // SiLU
        o[i] = (__bf16)s;
    }
    *(v4bf*)(Abf + e0) = o;
}

// ---- K2: xW = A(bf16) @ Wx^T(bf16) + b, f32 out.
//      Each wave: 32x32 output tile (2x2 WMMA tiles) over K=1024 -> 128 WMMA/wave,
//      fragments reused 2x each (1KB traffic per WMMA). ----
__global__ void k_gemm(const __bf16* __restrict__ Abf, const __bf16* __restrict__ Wxbf,
                       const float* __restrict__ bias, float* __restrict__ xw) {
    int wid  = blockIdx.x * (blockDim.x >> 5) + (threadIdx.x >> 5);  // 16384 wave jobs
    int lane = threadIdx.x & 31;
    int m0 = (wid >> 5) * 32;          // 512 M-blocks of 32 rows
    int n0 = (wid & 31) * 32;          // 32 N-blocks of 32 cols
    int row = lane & 15, hi = lane >> 4;
    const __bf16* a0row = Abf  + (long)(m0 + row) * Dn;
    const __bf16* a1row = a0row + 16L * Dn;
    const __bf16* b0row = Wxbf + (long)(n0 + row) * Dn + hi * 16;  // B[k,n]=Wx[n,k]: contiguous
    const __bf16* b1row = b0row + 16L * Dn;
    v8f c00 = {}, c01 = {}, c10 = {}, c11 = {};
#pragma unroll
    for (int kk = 0; kk < 32; ++kk) {
        v16bf a0 = load_a_frag(a0row + kk * 32, hi);
        v16bf a1 = load_a_frag(a1row + kk * 32, hi);
        v16bf b0 = *(const v16bf*)(b0row + kk * 32);
        v16bf b1 = *(const v16bf*)(b1row + kk * 32);
        c00 = __builtin_amdgcn_wmma_f32_16x16x32_bf16(false, a0, false, b0, (short)0, c00, false, false);
        c01 = __builtin_amdgcn_wmma_f32_16x16x32_bf16(false, a0, false, b1, (short)0, c01, false, false);
        c10 = __builtin_amdgcn_wmma_f32_16x16x32_bf16(false, a1, false, b0, (short)0, c10, false, false);
        c11 = __builtin_amdgcn_wmma_f32_16x16x32_bf16(false, a1, false, b1, (short)0, c11, false, false);
    }
    int nA = n0 + row, nB = n0 + 16 + row;
    float bbA = bias[nA], bbB = bias[nB];
#pragma unroll
    for (int r = 0; r < 8; ++r) {
        int mA = m0 + r + hi * 8;          // C/D layout: M = r + 8*hi, N = lane&15
        int mB = mA + 16;
        xw[(long)mA * Dn + nA] = c00[r] + bbA;
        xw[(long)mA * Dn + nB] = c01[r] + bbB;
        xw[(long)mB * Dn + nA] = c10[r] + bbA;
        xw[(long)mB * Dn + nB] = c11[r] + bbB;
    }
}

// ---- K3: persistent recurrence. 32 blocks x 64 thr (2 waves). Wave w owns columns w*16..+15.
//      Wh tile (16 contiguous rows = 32KB) staged to LDS via TDM; h double-buffered bf16
//      [16,1024] in global (L2-resident), one grid barrier per timestep. ----
__global__ void __launch_bounds__(64) k_rec(const float* __restrict__ xw, float* __restrict__ h_all,
                                            __bf16* __restrict__ hbuf, const __bf16* __restrict__ Whbf,
                                            unsigned* __restrict__ sync) {
    __shared__ __align__(32) __bf16 Bsh[2][16][Dn];       // 64 KB: per-wave row-major Wh tile
    int warp = threadIdx.x >> 5, lane = threadIdx.x & 31;
    int w = blockIdx.x * 2 + warp;
    int n0 = w * 16;
    int row = lane & 15, hi = lane >> 4;

    // Stage Wh rows n0..n0+15 (contiguous 32KB) into this wave's LDS half.
#ifdef HAVE_TDM
    tdm_load_1d(Whbf + (long)n0 * Dn, (unsigned)(warp * 16 * Dn * 2), 16u * Dn);
#if __has_builtin(__builtin_amdgcn_s_wait_tensorcnt)
    __builtin_amdgcn_s_wait_tensorcnt(0);
#endif
    // TDM wrote LDS behind the compiler's back: escape Bsh's address into an asm with a
    // memory clobber so the ds_loads below cannot be folded to undef / eliminated.
    {
        __bf16* esc = &Bsh[0][0][0];
        asm volatile("" : "+v"(esc) : : "memory");
    }
#else
    {
        const v8bf* src = (const v8bf*)(Whbf + (long)n0 * Dn);
        v8bf* dst = (v8bf*)&Bsh[warp][0][0];
        for (int i = lane; i < (16 * Dn) / 8; i += 32) dst[i] = src[i];
    }
#endif
    __syncthreads();

    int n = n0 + row;
    for (int t = 0; t < Tn; ++t) {
        const __bf16* hread  = hbuf + (t & 1) * (16 * Dn);
        __bf16*       hwrite = hbuf + ((t + 1) & 1) * (16 * Dn);
        v8f c = {};
#pragma unroll
        for (int kk = 0; kk < 32; ++kk) {
            v16bf a   = load_a_frag(hread + row * Dn + kk * 32, hi);
            v16bf bfr = *(const v16bf*)&Bsh[warp][row][kk * 32 + hi * 16];  // 32B contiguous
            c = __builtin_amdgcn_wmma_f32_16x16x32_bf16(false, a, false, bfr, (short)0, c, false, false);
        }
        if (hi == 0) {                          // lanes 0..15 hold batches 0..7 (M = r)
#pragma unroll
            for (int r = 0; r < 8; ++r) {
                long idx = ((long)r * Tn + t) * Dn + n;
                float v = tanhf(c[r] + xw[idx]);
                h_all[idx] = v;
                hwrite[r * Dn + n] = (__bf16)v;  // rows 8..15 stay zero forever
            }
        }
        grid_sync(&sync[0], &sync[1], gridDim.x);
    }
}

// ---- K4: output = h_all * silu(z); h_last = h_all[:, T-1]. 4 elems/thread, overwrites xw. ----
__global__ void k_final(const float* __restrict__ z, const float* __restrict__ h_all,
                        float* __restrict__ out2, float* __restrict__ h_last) {
    long e0 = ((long)blockIdx.x * blockDim.x + threadIdx.x) * 4;
    v4f h4 = *(const v4f*)(h_all + e0);
    v4f z4 = *(const v4f*)(z + e0);
    v4f o;
#pragma unroll
    for (int i = 0; i < 4; ++i) o[i] = h4[i] * (z4[i] / (1.0f + __expf(-z4[i])));
    *(v4f*)(out2 + e0) = o;
    long td = e0 % ((long)Tn * Dn);
    if (td >= (long)(Tn - 1) * Dn) {
        long b_ = e0 / ((long)Tn * Dn);
        int  d  = (int)(td - (long)(Tn - 1) * Dn);
        *(v4f*)(h_last + b_ * Dn + d) = h4;
    }
}

extern "C" void kernel_launch(void* const* d_in, const int* in_sizes, int n_in,
                              void* d_out, int out_size, void* d_ws, size_t ws_size,
                              hipStream_t stream) {
    (void)in_sizes; (void)n_in; (void)out_size; (void)ws_size;
    const float* x      = (const float*)d_in[0];
    const float* z      = (const float*)d_in[1];
    const float* h_init = (const float*)d_in[2];
    const float* W_x    = (const float*)d_in[3];
    const float* W_h    = (const float*)d_in[4];
    const float* b      = (const float*)d_in[5];
    const float* cw     = (const float*)d_in[6];
    const float* cb     = (const float*)d_in[7];

    float* out    = (float*)d_out;
    float* h_all  = out;                                  // region 1: [B,T,D]
    float* xw_out = out + (long)Bn * Tn * Dn;             // region 2: xw scratch, then final output
    float* h_last = out + 2L * Bn * Tn * Dn;              // region 3: [B,D]

    __bf16* Abf  = (__bf16*)d_ws;                         // B*T*D bf16   (32 MB)
    __bf16* Wxbf = Abf + (long)Bn * Tn * Dn;              // D*D bf16     (2 MB)
    __bf16* Whbf = Wxbf + (long)Dn * Dn;                  // D*D bf16     (2 MB)
    __bf16* hbuf = Whbf + (long)Dn * Dn;                  // 2*16*D bf16  (64 KB)
    unsigned* sync = (unsigned*)(hbuf + 2 * 16 * Dn);     // 2 words

    k_setup<<<4096, 256, 0, stream>>>(W_x, W_h, h_init, Wxbf, Whbf, hbuf, sync);
    k_conv <<<16384, 256, 0, stream>>>(x, cw, cb, Abf);
    k_gemm <<<2048, 256, 0, stream>>>(Abf, Wxbf, b, xw_out);
    k_rec  <<<32, 64, 0, stream>>>(xw_out, h_all, hbuf, Whbf, sync);
    k_final<<<16384, 256, 0, stream>>>(z, h_all, xw_out, h_last);
}
